// Local_Attention_21131239096481
// MI455X (gfx1250) — compile-verified, matrix operand order checked by reading, not run on hardware
//
#include <hip/hip_runtime.h>
#include <hip/hip_bf16.h>
#include <math.h>

// ---------------------------------------------------------------------------
// Types for CDNA5 WMMA (wave32): A/B = v16h (8 VGPRs), C/D = v8f (8 VGPRs)
// ---------------------------------------------------------------------------
typedef _Float16 v16h __attribute__((ext_vector_type(16)));
typedef _Float16 v8h  __attribute__((ext_vector_type(8)));
typedef float    v8f  __attribute__((ext_vector_type(8)));
typedef float    v4f  __attribute__((ext_vector_type(4)));
typedef unsigned int u32x4 __attribute__((ext_vector_type(4)));
typedef int          i32x8 __attribute__((ext_vector_type(8)));
typedef int          i32x4 __attribute__((ext_vector_type(4)));

#define CDIM 128            // C_Q = C_KV = H*C = 128
#define WELEM (CDIM * CDIM) // elements per weight matrix
#define INV_SQRT_C 0.17677669529663687f   // 1/sqrt(32)

// ---------------------------------------------------------------------------
// TDM: DMA one 128x128 f16 tile (32 KB) from global into LDS.
// D# per ISA 8.3/8.4: group0 = {count=1, lds_addr, global_addr, type=2},
// group1 = {data_size=2B, tensor_dim0/1=128, tile_dim0/1=128, stride0=128}.
// Issued by one wave (EXEC ignored -> one DMA per issuing wave); completion
// via TENSORcnt (s_wait_tensorcnt) then workgroup barrier.
// This toolchain uses the 6-arg builtin form (extra i32x8 group arg).
// ---------------------------------------------------------------------------
__device__ __forceinline__ void tdm_load_w(_Float16* lds_dst, const _Float16* gsrc) {
  const unsigned lds = (unsigned)(uintptr_t)lds_dst;        // addr[31:0] = LDS offset
  const unsigned long long ga = (unsigned long long)(uintptr_t)gsrc;
  u32x4 g0;
  g0[0] = 1u;                                               // count=1 (user mode)
  g0[1] = lds;                                              // lds_addr (bytes)
  g0[2] = (unsigned)ga;                                     // global_addr[31:0]
  g0[3] = (unsigned)((ga >> 32) & 0x01FFFFFFu) | (2u << 30);// addr[56:32] | type=2
  i32x8 g1;
  g1[0] = 0x00010000;   // workgroup_mask=0, data_size=1 (2 bytes)
  g1[1] = 0x00800000;   // tensor_dim0 = 128  (bits[31:16] = lo16)
  g1[2] = 0x00800000;   // tensor_dim1 = 128  (bits[31:16] = lo16)
  g1[3] = 0x00800000;   // tile_dim0   = 128  (bits[31:16])
  g1[4] = 0x00000080;   // tile_dim1   = 128  (bits[15:0]); tile_dim2 = 0
  g1[5] = 128;          // tensor_dim0_stride = 128 (lo32)
  g1[6] = 0;            // stride0 hi / dim1_stride lo
  g1[7] = 0;
  const i32x4 z4 = {0, 0, 0, 0};                            // groups 2/3 unused (2D)
  const i32x8 z8 = {0, 0, 0, 0, 0, 0, 0, 0};
  __builtin_amdgcn_tensor_load_to_lds(g0, g1, z4, z4, z8, 0);
}

// ---------------------------------------------------------------------------
// Fragment loaders (16-bit A layout, ISA 7.12.2): lane L<16 = row M=L, halves
// [0..7]=K(k0..k0+7), [8..15]=K(k0+16..k0+23); lanes 16..31 shifted by K+8.
// B mirrors this with columns per lane (weights stored pre-transposed).
// ---------------------------------------------------------------------------
__device__ __forceinline__ v16h cvt_frag(v4f a, v4f b, v4f c, v4f d) {
  v16h r;
#pragma unroll
  for (int i = 0; i < 4; ++i) {
    r[i]      = (_Float16)a[i];
    r[4 + i]  = (_Float16)b[i];
    r[8 + i]  = (_Float16)c[i];
    r[12 + i] = (_Float16)d[i];
  }
  return r;
}

__device__ __forceinline__ v16h load_a_frag_global(const float* rowptr, int kblock, int lane) {
  const int kb = kblock + ((lane >> 4) << 3);
  const v4f* p0 = reinterpret_cast<const v4f*>(rowptr + kb);
  const v4f* p1 = reinterpret_cast<const v4f*>(rowptr + kb + 16);
  return cvt_frag(p0[0], p0[1], p1[0], p1[1]);
}

__device__ __forceinline__ v16h load_frag_lds(const _Float16* base, int kblock, int lane) {
  const int kb = kblock + ((lane >> 4) << 3);
  const v8h lo = *reinterpret_cast<const v8h*>(base + kb);
  const v8h hi = *reinterpret_cast<const v8h*>(base + kb + 16);
  v16h r;
#pragma unroll
  for (int i = 0; i < 8; ++i) { r[i] = lo[i]; r[8 + i] = hi[i]; }
  return r;
}

// One wave: 16 rows x 128 cols, K=128 -> 8 C/D tiles, 32 WMMAs.
__device__ __forceinline__ void gemm_tiles(const v16h* a, const _Float16* wt, int lane,
                                           const float* __restrict__ bias, float scale,
                                           v8f* acc) {
  const int nsub = lane & 15;
#pragma unroll
  for (int nt = 0; nt < 8; ++nt) {
    const _Float16* col = wt + (nt * 16 + nsub) * CDIM;
    v8f c = {0.f, 0.f, 0.f, 0.f, 0.f, 0.f, 0.f, 0.f};
#pragma unroll
    for (int t = 0; t < 4; ++t) {
      v16h b = load_frag_lds(col, t * 32, lane);
      c = __builtin_amdgcn_wmma_f32_16x16x32_f16(false, a[t], false, b,
                                                 (short)0, c, false, false);
    }
    const float bb = bias ? bias[nt * 16 + nsub] : 0.0f;
#pragma unroll
    for (int i = 0; i < 8; ++i) c[i] = (c[i] + bb) * scale;
    acc[nt] = c;
  }
}

// ---------------------------------------------------------------------------
// One-shot: convert all 5 weight matrices to transposed f16 LDS image
// (wt16[m][n*128+k] = W_m[k][n]); 160 KB, L2-resident for the whole run.
// ---------------------------------------------------------------------------
__global__ void prep_weights_kernel(const float* __restrict__ Wq, const float* __restrict__ Wk,
                                    const float* __restrict__ Wv, const float* __restrict__ Wg,
                                    const float* __restrict__ Wo, _Float16* __restrict__ wt16) {
  const int g = blockIdx.x * 256 + threadIdx.x;
  if (g >= 5 * WELEM) return;
  const int m = g >> 14;                 // matrix id
  const int i = g & (WELEM - 1);         // n*128 + k
  const int nn = i >> 7, k = i & 127;
  const float* W = (m == 0) ? Wq : (m == 1) ? Wk : (m == 2) ? Wv : (m == 3) ? Wg : Wo;
  wt16[g] = (_Float16)W[k * CDIM + nn];
}

// ---------------------------------------------------------------------------
// Kernel 1: fused Q/K/V projections. Weights arrive via TDM, double-buffered
// so the next DMA overlaps the current GEMM. Q,K stay in registers; the score
// s = (q.k)/sqrt(C) is reduced with wave32 shfl_xor. V spilled to ws.
// ---------------------------------------------------------------------------
__global__ __launch_bounds__(256, 1) void fused_proj_kernel(
    const float* __restrict__ q_x, const float* __restrict__ kv_x,
    const _Float16* __restrict__ wt16, const float* __restrict__ bq,
    float* __restrict__ s_out, float* __restrict__ v_out) {
  __shared__ _Float16 wbuf0[WELEM];   // 32 KB
  __shared__ _Float16 wbuf1[WELEM];   // 32 KB
  const int tid  = threadIdx.x;
  const int lane = tid & 31;
  const int wave = tid >> 5;
  const size_t m0 = (size_t)blockIdx.x * 128 + (size_t)wave * 16;

  // Kick off Wq and prefetch Wk; wait only for Wq (in-order per-wave TDM).
  if (wave == 0) {
    tdm_load_w(wbuf0, wt16 + 0 * WELEM);    // Wq^T
    tdm_load_w(wbuf1, wt16 + 1 * WELEM);    // Wk^T (prefetch)
    __builtin_amdgcn_s_wait_tensorcnt(1);
  }
  __syncthreads();

  const float* qrow = q_x + (m0 + (size_t)(lane & 15)) * CDIM;
  v16h aq[4];
#pragma unroll
  for (int t = 0; t < 4; ++t) aq[t] = load_a_frag_global(qrow, t * 32, lane);
  v8f qacc[8];
  gemm_tiles(aq, wbuf0, lane, bq, INV_SQRT_C, qacc);  // q = (x@Wq + bq)/sqrt(C)

  __syncthreads();                                    // wbuf0 reads done
  if (wave == 0) {
    tdm_load_w(wbuf0, wt16 + 2 * WELEM);              // Wv^T (prefetch into buf0)
    __builtin_amdgcn_s_wait_tensorcnt(1);             // ensure Wk complete
  }
  __syncthreads();

  const float* krow = kv_x + (m0 + (size_t)(lane & 15)) * CDIM;
  v16h akv[4];
#pragma unroll
  for (int t = 0; t < 4; ++t) akv[t] = load_a_frag_global(krow, t * 32, lane);
  v8f kacc[8];
  gemm_tiles(akv, wbuf1, lane, nullptr, 1.0f, kacc);

  // s[row][h] = sum_c q*k over head h (tiles 2h,2h+1), 16-lane xor reduction.
#pragma unroll
  for (int r = 0; r < 8; ++r) {
#pragma unroll
    for (int h = 0; h < 4; ++h) {
      float p = qacc[2 * h][r] * kacc[2 * h][r] + qacc[2 * h + 1][r] * kacc[2 * h + 1][r];
      p += __shfl_xor(p, 1); p += __shfl_xor(p, 2);
      p += __shfl_xor(p, 4); p += __shfl_xor(p, 8);
      if ((lane & 15) == h) {
        const size_t row = m0 + (size_t)r + (size_t)((lane >> 4) << 3);
        s_out[row * 4 + h] = p * INV_SQRT_C;
      }
    }
  }

  __syncthreads();
  if (wave == 0) __builtin_amdgcn_s_wait_tensorcnt(0);  // Wv landed
  __syncthreads();

  v8f vacc[8];
  gemm_tiles(akv, wbuf0, lane, nullptr, 1.0f, vacc);
#pragma unroll
  for (int nt = 0; nt < 8; ++nt) {
    const int col = nt * 16 + (lane & 15);
#pragma unroll
    for (int r = 0; r < 8; ++r) {
      const size_t row = m0 + (size_t)r + (size_t)((lane >> 4) << 3);
      v_out[row * CDIM + col] = vacc[nt][r];
    }
  }
}

// ---------------------------------------------------------------------------
// Scatter-softmax pipeline (exact fp32, native f32 atomics on gfx1250).
// ---------------------------------------------------------------------------
__global__ void init_ws_kernel(float* __restrict__ segmax, float* __restrict__ denom,
                               float* __restrict__ outacc, int n) {
  const int i = blockIdx.x * blockDim.x + threadIdx.x;
  if (i < n * CDIM) outacc[i] = 0.0f;
  if (i < n * 4) { segmax[i] = -INFINITY; denom[i] = 0.0f; }
}

__global__ void seg_max_kernel(const float* __restrict__ s, const int* __restrict__ idx,
                               float* __restrict__ segmax, int n) {
  const int i = blockIdx.x * blockDim.x + threadIdx.x;
  if (i >= n * 4) return;
  const int a = i >> 2, h = i & 3;
  const int t = idx[a];
  __hip_atomic_fetch_max(&segmax[t * 4 + h], s[i],
                         __ATOMIC_RELAXED, __HIP_MEMORY_SCOPE_AGENT);
}

__global__ void seg_expsum_kernel(const float* __restrict__ s, const int* __restrict__ idx,
                                  const float* __restrict__ segmax, float* __restrict__ e,
                                  float* __restrict__ denom, int n) {
  const int i = blockIdx.x * blockDim.x + threadIdx.x;
  if (i >= n * 4) return;
  const int a = i >> 2, h = i & 3;
  const int t = idx[a];
  const float ev = __expf(s[i] - segmax[t * 4 + h]);
  e[i] = ev;
  __hip_atomic_fetch_add(&denom[t * 4 + h], ev,
                         __ATOMIC_RELAXED, __HIP_MEMORY_SCOPE_AGENT);
}

// grid = n atoms, 128 threads = one per output channel
__global__ void scatter_av_kernel(const float* __restrict__ e, const float* __restrict__ denom,
                                  const float* __restrict__ v, const int* __restrict__ idx,
                                  float* __restrict__ outacc) {
  const int a = blockIdx.x;
  const int c = threadIdx.x;
  const int h = c >> 5;
  const int t = idx[a];
  const float alpha = e[a * 4 + h] / denom[t * 4 + h];
  __hip_atomic_fetch_add(&outacc[(size_t)t * CDIM + c], alpha * v[(size_t)a * CDIM + c],
                         __ATOMIC_RELAXED, __HIP_MEMORY_SCOPE_AGENT);
}

// ---------------------------------------------------------------------------
// Kernel 6: gate = sigmoid(q_x@Wg); X = outacc*gate routed through LDS
// (C/D layout -> A layout); out = X@Wo + bo. Wo DMA overlaps gating math.
// ---------------------------------------------------------------------------
__global__ __launch_bounds__(256, 1) void gate_out_kernel(
    const float* __restrict__ q_x, const _Float16* __restrict__ wt16,
    const float* __restrict__ bo, const float* __restrict__ outacc,
    float* __restrict__ out) {
  __shared__ _Float16 wt[WELEM];   // Wg^T, then Wo^T (32 KB)
  __shared__ _Float16 xt[WELEM];   // gated X rows as f16 (32 KB)
  const int tid  = threadIdx.x;
  const int lane = tid & 31;
  const int wave = tid >> 5;
  const size_t m0 = (size_t)blockIdx.x * 128 + (size_t)wave * 16;

  if (wave == 0) {
    tdm_load_w(wt, wt16 + 3 * WELEM);           // Wg^T
    __builtin_amdgcn_s_wait_tensorcnt(0);
  }
  __syncthreads();

  const float* qrow = q_x + (m0 + (size_t)(lane & 15)) * CDIM;
  v16h aq[4];
#pragma unroll
  for (int t = 0; t < 4; ++t) aq[t] = load_a_frag_global(qrow, t * 32, lane);
  v8f gacc[8];
  gemm_tiles(aq, wt, lane, nullptr, 1.0f, gacc);

  __syncthreads();                              // Wg reads done
  if (wave == 0) tdm_load_w(wt, wt16 + 4 * WELEM);  // Wo^T DMA overlaps gating

  // X = outacc * sigmoid(gacc) -> LDS row-major f16 (C/D -> A layout bridge)
#pragma unroll
  for (int nt = 0; nt < 8; ++nt) {
    const int col = nt * 16 + (lane & 15);
#pragma unroll
    for (int r = 0; r < 8; ++r) {
      const int rloc = wave * 16 + r + ((lane >> 4) << 3);
      const size_t row = (size_t)blockIdx.x * 128 + rloc;
      const float g = 1.0f / (1.0f + __expf(-gacc[nt][r]));
      xt[rloc * CDIM + col] = (_Float16)(outacc[row * CDIM + col] * g);
    }
  }
  __syncthreads();
  if (wave == 0) __builtin_amdgcn_s_wait_tensorcnt(0);
  __syncthreads();

  const _Float16* xrow = xt + (wave * 16 + (lane & 15)) * CDIM;
  v16h ax[4];
#pragma unroll
  for (int t = 0; t < 4; ++t) ax[t] = load_frag_lds(xrow, t * 32, lane);
  v8f oacc[8];
  gemm_tiles(ax, wt, lane, bo, 1.0f, oacc);     // X@Wo + bo

#pragma unroll
  for (int nt = 0; nt < 8; ++nt) {
    const int col = nt * 16 + (lane & 15);
#pragma unroll
    for (int r = 0; r < 8; ++r) {
      const size_t row = m0 + (size_t)r + (size_t)((lane >> 4) << 3);
      out[row * CDIM + col] = oacc[nt][r];
    }
  }
}

// ---------------------------------------------------------------------------
// Host launch
// ---------------------------------------------------------------------------
extern "C" void kernel_launch(void* const* d_in, const int* in_sizes, int n_in,
                              void* d_out, int out_size, void* d_ws, size_t ws_size,
                              hipStream_t stream) {
  const float* q_x  = (const float*)d_in[0];
  const float* kv_x = (const float*)d_in[1];
  const int*   idx  = (const int*)d_in[2];
  const float* Wq   = (const float*)d_in[3];
  const float* bq   = (const float*)d_in[4];
  const float* Wk   = (const float*)d_in[5];
  const float* Wv   = (const float*)d_in[6];
  const float* Wg   = (const float*)d_in[7];
  const float* Wo   = (const float*)d_in[8];
  const float* bo   = (const float*)d_in[9];
  const int n = in_sizes[2];          // atoms (262144, multiple of 128)

  // Workspace: s[n*4] e[n*4] segmax[n*4] denom[n*4] V[n*128] outacc[n*128]
  //            wt16[5*128*128 f16]  (~285 MB total)
  float* ws     = (float*)d_ws;
  float* s      = ws;
  float* e      = s      + (size_t)n * 4;
  float* segmax = e      + (size_t)n * 4;
  float* denom  = segmax + (size_t)n * 4;
  float* vbuf   = denom  + (size_t)n * 4;
  float* outacc = vbuf   + (size_t)n * CDIM;
  _Float16* wt16 = (_Float16*)(outacc + (size_t)n * CDIM);
  float* out    = (float*)d_out;

  const int nblk = n / 128;

  prep_weights_kernel<<<(5 * WELEM + 255) / 256, 256, 0, stream>>>(Wq, Wk, Wv, Wg, Wo, wt16);
  init_ws_kernel<<<(n * CDIM + 255) / 256, 256, 0, stream>>>(segmax, denom, outacc, n);
  fused_proj_kernel<<<nblk, 256, 0, stream>>>(q_x, kv_x, wt16, bq, s, vbuf);
  seg_max_kernel<<<(n * 4 + 255) / 256, 256, 0, stream>>>(s, idx, segmax, n);
  seg_expsum_kernel<<<(n * 4 + 255) / 256, 256, 0, stream>>>(s, idx, segmax, e, denom, n);
  scatter_av_kernel<<<n, CDIM, 0, stream>>>(e, denom, vbuf, idx, outacc);
  gate_out_kernel<<<nblk, 256, 0, stream>>>(q_x, wt16, bo, outacc, out);
}
